// Simple_Model_26414048870998
// MI455X (gfx1250) — matrix operand order repaired and checked by reading, not compiled, not run on hardware
//
#include <hip/hip_runtime.h>

#define N_NODES 20000
#define N_EDGES 320000
#define NHEADS  8
#define NHID_   64
#define HC_     512
#define HC2_    1024
#define NLABEL_ 40

typedef __attribute__((ext_vector_type(16))) __bf16       v16bf;
typedef __attribute__((ext_vector_type(8)))  float        v8f;
typedef __attribute__((ext_vector_type(4)))  unsigned int v4u;

// ---------- helpers ----------
static __device__ __forceinline__ unsigned short f2bf(float f) {
  unsigned int u = __float_as_uint(f);
  u += 0x7FFFu + ((u >> 16) & 1u);          // round-to-nearest-even
  return (unsigned short)(u >> 16);
}
// order-preserving float<->uint encoding for atomicMax on floats
static __device__ __forceinline__ unsigned int encf(float f) {
  unsigned int u = __float_as_uint(f);
  return (u & 0x80000000u) ? ~u : (u | 0x80000000u);
}
static __device__ __forceinline__ float decf(unsigned int e) {
  unsigned int u = (e & 0x80000000u) ? (e ^ 0x80000000u) : ~e;
  return __uint_as_float(u);
}

// ---------- elementwise utility kernels ----------
__global__ void k_fill_u32(unsigned int* p, unsigned int v, int n) {
  int i = blockIdx.x * blockDim.x + threadIdx.x;
  if (i < n) p[i] = v;
}
__global__ void k_fill_f32(float* p, float v, int n) {
  int i = blockIdx.x * blockDim.x + threadIdx.x;
  if (i < n) p[i] = v;
}
__global__ void k_f32_to_bf16(unsigned short* d, const float* __restrict__ s, int n) {
  int i = blockIdx.x * blockDim.x + threadIdx.x;
  if (i < n) d[i] = f2bf(s[i]);
}
// concat [agg | h] -> bf16 [N, 2*HC]
__global__ void k_concat_bf16(const float* __restrict__ agg, const float* __restrict__ h,
                              unsigned short* __restrict__ out) {
  long i = (long)blockIdx.x * blockDim.x + threadIdx.x;
  if (i >= (long)N_NODES * HC2_) return;
  int n = (int)(i >> 10);
  int c = (int)(i & 1023);
  float v = (c < HC_) ? agg[(long)n * HC_ + c] : h[(long)n * HC_ + (c - HC_)];
  out[i] = f2bf(v);
}

// ---------- pack weight W[K,Nc] (f32, row-major) into WMMA B-fragment layout ----------
// layout: dword index = ((kt*Nt16 + nt)*32 + lane)*8 + v
//   n = nt*16 + lane%16 ; k = kt*32 + (v/4)*16 + (lane/16)*8 + 2*(v%4)  (pair k, k+1 packed lo|hi)
__global__ void k_pack_b(unsigned int* __restrict__ dst, const float* __restrict__ W,
                         int K, int Nc, int Nt16) {
  int idx = blockIdx.x * blockDim.x + threadIdx.x;
  int total = (K >> 5) * Nt16 * 256;
  if (idx >= total) return;
  int v    = idx & 7;
  int lane = (idx >> 3) & 31;
  int tile = idx >> 8;
  int nt   = tile % Nt16;
  int kt   = tile / Nt16;
  int n = nt * 16 + (lane & 15);
  int k = kt * 32 + (v >> 2) * 16 + (lane >> 4) * 8 + 2 * (v & 3);
  float lo = (n < Nc) ? W[(long)k * Nc + n] : 0.f;
  float hi = (n < Nc) ? W[(long)(k + 1) * Nc + n] : 0.f;
  dst[idx] = (unsigned int)f2bf(lo) | ((unsigned int)f2bf(hi) << 16);
}

// ---------- WMMA GEMM: C[M,Nc] = A(bf16,[M,K]) x Bpacked + bias, activation ----------
// act: 0=none, 1=GELU(exact erf), 2=ELU
__global__ __launch_bounds__(256)
void k_gemm_bf16(const unsigned short* __restrict__ A, const unsigned int* __restrict__ Bp,
                 float* __restrict__ C, int M, int K, int Nc, int Nt16,
                 const float* __restrict__ bias1, const float* __restrict__ bias2, int act)
{
  const int lane  = threadIdx.x & 31;
  const int wave  = threadIdx.x >> 5;
  const int m0    = blockIdx.x * 32 + (wave & 1) * 16;
  const int ntile = blockIdx.y * 4 + (wave >> 1);
  if (ntile >= Nt16 || m0 >= M) return;

  const int half = lane >> 4;           // K-half selector (ISA A/B fragment layout)
  const int mrow = m0 + (lane & 15);
  const unsigned short* aRow  = A + (long)mrow * K + half * 8;
  const unsigned int*   bBase = Bp + ((long)ntile * 32 + lane) * 8;
  const long bKStride = (long)Nt16 * 256;

  v8f acc = {};
  union { v4u u[2]; v16bf bf; } fa, fb;

  const int ktiles = K >> 5;
  for (int kt = 0; kt < ktiles; ++kt) {
    const unsigned short* ap = aRow + kt * 32;
    fa.u[0] = *(const v4u*)(ap);          // A[m, kk+half*8 .. +7]
    fa.u[1] = *(const v4u*)(ap + 16);     // A[m, kk+16+half*8 .. +7]
    const unsigned int* bp = bBase + kt * bKStride;
    fb.u[0] = *(const v4u*)(bp);
    fb.u[1] = *(const v4u*)(bp + 4);
    acc = __builtin_amdgcn_wmma_f32_16x16x32_bf16(
        false, fa.bf, false, fb.bf, (short)0, acc, false, false);
  }

  const int col = ntile * 16 + (lane & 15);
  if (col < Nc) {
    float b = bias1 ? bias1[col] : 0.f;
    if (bias2) b += bias2[col];
    #pragma unroll
    for (int r = 0; r < 8; ++r) {
      int row = m0 + half * 8 + r;        // C/D VGPR layout: lanes16-31 hold M=8..15
      float v = acc[r] + b;
      if (act == 1)      v = 0.5f * v * (1.0f + erff(v * 0.70710678118654752f));
      else if (act == 2) v = (v > 0.f) ? v : (expf(v) - 1.0f);
      C[(long)row * Nc + col] = v;
    }
  }
}

// ---------- attention score kernels ----------
__global__ void k_att_scores(const float* __restrict__ h, const float* __restrict__ attS,
                             const float* __restrict__ attD,
                             float* __restrict__ aS, float* __restrict__ aD) {
  int i = blockIdx.x * blockDim.x + threadIdx.x;   // n*NHEADS + head
  if (i >= N_NODES * NHEADS) return;
  int n = i >> 3, hd = i & 7;
  const float4* hp = (const float4*)(h + (long)n * HC_ + hd * NHID_);
  const float4* sp = (const float4*)(attS + hd * NHID_);
  const float4* dp = (const float4*)(attD + hd * NHID_);
  float s = 0.f, d = 0.f;
  #pragma unroll
  for (int j = 0; j < NHID_ / 4; ++j) {
    float4 hv = hp[j], sv = sp[j], dv = dp[j];
    s += hv.x * sv.x + hv.y * sv.y + hv.z * sv.z + hv.w * sv.w;
    d += hv.x * dv.x + hv.y * dv.y + hv.z * dv.z + hv.w * dv.w;
  }
  aS[i] = s; aD[i] = d;
}

__global__ void k_edge_scores(const int* __restrict__ src, const int* __restrict__ dst,
                              const float* __restrict__ aS, const float* __restrict__ aD,
                              float* __restrict__ es, unsigned int* __restrict__ emax) {
  int i = blockIdx.x * blockDim.x + threadIdx.x;   // e*NHEADS + head
  if (i >= N_EDGES * NHEADS) return;
  int e = i >> 3, hd = i & 7;
  int s = src[e], d = dst[e];
  float v = aS[s * NHEADS + hd] + aD[d * NHEADS + hd];
  v = (v > 0.f) ? v : 0.2f * v;                    // leaky relu
  es[i] = v;
  atomicMax(&emax[d * NHEADS + hd], encf(v));
}

__global__ void k_edge_exp(const int* __restrict__ dst, const unsigned int* __restrict__ emax,
                           float* __restrict__ es, float* __restrict__ denom) {
  int i = blockIdx.x * blockDim.x + threadIdx.x;
  if (i >= N_EDGES * NHEADS) return;
  int e = i >> 3, hd = i & 7;
  int d = dst[e];
  float m  = decf(emax[d * NHEADS + hd]);
  float ee = expf(es[i] - m);
  es[i] = ee;
  atomicAdd(&denom[d * NHEADS + hd], ee);
}

// one wave per edge; lane covers 16 contiguous channels (within one head: head = lane/4)
__global__ void k_edge_agg(const int* __restrict__ src, const int* __restrict__ dst,
                           const float* __restrict__ es, const float* __restrict__ denom,
                           const float* __restrict__ h, float* __restrict__ agg) {
  int gid = blockIdx.x * blockDim.x + threadIdx.x;
  int e = gid >> 5;
  if (e >= N_EDGES) return;
  int lane = gid & 31;
  int s = src[e], d = dst[e];
  int hd = lane >> 2;
  float alpha = es[e * NHEADS + hd] / (denom[d * NHEADS + hd] + 1e-16f);
  int c0 = lane * 16;
  const float4* hp = (const float4*)(h + (long)s * HC_ + c0);
  float* ap = agg + (long)d * HC_ + c0;
  #pragma unroll
  for (int j = 0; j < 4; ++j) {
    float4 hv = hp[j];
    atomicAdd(ap + 4 * j + 0, alpha * hv.x);
    atomicAdd(ap + 4 * j + 1, alpha * hv.y);
    atomicAdd(ap + 4 * j + 2, alpha * hv.z);
    atomicAdd(ap + 4 * j + 3, alpha * hv.w);
  }
}

// ---------- host ----------
static inline int cdiv(long a, int b) { return (int)((a + b - 1) / b); }

extern "C" void kernel_launch(void* const* d_in, const int* in_sizes, int n_in,
                              void* d_out, int out_size, void* d_ws, size_t ws_size,
                              hipStream_t stream) {
  const float* x    = (const float*)d_in[0];
  const int*   eidx = (const int*)d_in[1];
  const int*   src  = eidx;
  const int*   dst  = eidx + N_EDGES;

  const float* g_W[2]    = { (const float*)d_in[2],  (const float*)d_in[10] };
  const float* g_aS[2]   = { (const float*)d_in[3],  (const float*)d_in[11] };
  const float* g_aD[2]   = { (const float*)d_in[4],  (const float*)d_in[12] };
  const float* g_pW1[2]  = { (const float*)d_in[5],  (const float*)d_in[13] };
  const float* g_pb1[2]  = { (const float*)d_in[6],  (const float*)d_in[14] };
  const float* g_pW2[2]  = { (const float*)d_in[7],  (const float*)d_in[15] };
  const float* g_pb2[2]  = { (const float*)d_in[8],  (const float*)d_in[16] };
  const float* g_bias[2] = { (const float*)d_in[9],  (const float*)d_in[17] };
  const float* f_W1 = (const float*)d_in[18];
  const float* f_b1 = (const float*)d_in[19];
  const float* f_W2 = (const float*)d_in[20];
  const float* f_b2 = (const float*)d_in[21];

  // ---- carve workspace ----
  char* wp = (char*)d_ws;
  auto carve = [&](size_t bytes) -> void* {
    void* p = (void*)wp;
    wp += (bytes + 255) & ~(size_t)255;
    return p;
  };
  float* h_buf   = (float*)carve((size_t)N_NODES * HC_  * 4);
  float* agg_buf = (float*)carve((size_t)N_NODES * HC_  * 4);
  float* t1_buf  = (float*)carve((size_t)N_NODES * HC2_ * 4);   // phi hidden / final hidden
  float* out_buf = (float*)carve((size_t)N_NODES * HC_  * 4);
  unsigned short* bfIn = (unsigned short*)carve((size_t)N_NODES * HC2_ * 2);
  unsigned short* bfT  = (unsigned short*)carve((size_t)N_NODES * HC_  * 2);
  float*        aS    = (float*)carve((size_t)N_NODES * NHEADS * 4);
  float*        aD    = (float*)carve((size_t)N_NODES * NHEADS * 4);
  float*        denom = (float*)carve((size_t)N_NODES * NHEADS * 4);
  unsigned int* emax  = (unsigned int*)carve((size_t)N_NODES * NHEADS * 4);
  float*        es    = (float*)carve((size_t)N_EDGES * NHEADS * 4);

  // packed weights: dwords = (K/32)*Nt16*256
  auto packAlloc = [&](int K, int Nt16) -> unsigned int* {
    return (unsigned int*)carve((size_t)(K >> 5) * Nt16 * 256 * 4);
  };
  unsigned int* p_gW[2]  = { packAlloc(256, 32),  packAlloc(512, 32) };
  unsigned int* p_pW1[2] = { packAlloc(1024, 32), packAlloc(1024, 32) };
  unsigned int* p_pW2[2] = { packAlloc(512, 32),  packAlloc(512, 32) };
  unsigned int* p_fW1    = packAlloc(512, 64);
  unsigned int* p_fW2    = packAlloc(1024, 3);

  auto pack = [&](unsigned int* dstp, const float* W, int K, int Nc, int Nt16) {
    int total = (K >> 5) * Nt16 * 256;
    k_pack_b<<<cdiv(total, 256), 256, 0, stream>>>(dstp, W, K, Nc, Nt16);
  };
  pack(p_gW[0],  g_W[0],   256,  512,  32);
  pack(p_gW[1],  g_W[1],   512,  512,  32);
  pack(p_pW1[0], g_pW1[0], 1024, 512,  32);
  pack(p_pW1[1], g_pW1[1], 1024, 512,  32);
  pack(p_pW2[0], g_pW2[0], 512,  512,  32);
  pack(p_pW2[1], g_pW2[1], 512,  512,  32);
  pack(p_fW1,    f_W1,     512,  1024, 64);
  pack(p_fW2,    f_W2,     1024, 40,   3);

  auto gemm = [&](const unsigned short* A, const unsigned int* Bp, float* Cout,
                  int K, int Nc, int Nt16, const float* b1, const float* b2, int act) {
    dim3 grid(N_NODES / 32, (Nt16 + 3) / 4);
    k_gemm_bf16<<<grid, 256, 0, stream>>>(A, Bp, Cout, N_NODES, K, Nc, Nt16, b1, b2, act);
  };

  const unsigned int ENC_NEG_INF = 0x007FFFFFu;  // encf(-inf)
  const int EH = N_EDGES * NHEADS;
  const int NH = N_NODES * NHEADS;

  // x -> bf16 input
  k_f32_to_bf16<<<cdiv((long)N_NODES * 256, 256), 256, 0, stream>>>(bfIn, x, N_NODES * 256);

  for (int L = 0; L < 2; ++L) {
    int Kin = (L == 0) ? 256 : 512;
    // h = x @ W
    gemm(bfIn, p_gW[L], h_buf, Kin, HC_, 32, nullptr, nullptr, 0);
    // attention logits per (node, head)
    k_att_scores<<<cdiv(NH, 256), 256, 0, stream>>>(h_buf, g_aS[L], g_aD[L], aS, aD);
    // init segment buffers
    k_fill_u32<<<cdiv(NH, 256), 256, 0, stream>>>(emax, ENC_NEG_INF, NH);
    k_fill_f32<<<cdiv(NH, 256), 256, 0, stream>>>(denom, 0.f, NH);
    k_fill_f32<<<cdiv((long)N_NODES * HC_, 256), 256, 0, stream>>>(agg_buf, 0.f, N_NODES * HC_);
    // softmax over edges grouped by dst
    k_edge_scores<<<cdiv(EH, 256), 256, 0, stream>>>(src, dst, aS, aD, es, emax);
    k_edge_exp<<<cdiv(EH, 256), 256, 0, stream>>>(dst, emax, es, denom);
    // agg[dst] += alpha * h[src]
    k_edge_agg<<<cdiv((long)N_EDGES * 32, 256), 256, 0, stream>>>(src, dst, es, denom, h_buf, agg_buf);
    // emb = concat(agg, h) -> bf16
    k_concat_bf16<<<cdiv((long)N_NODES * HC2_, 256), 256, 0, stream>>>(agg_buf, h_buf, bfIn);
    // t1 = gelu(emb @ phi_W1 + b1)
    gemm(bfIn, p_pW1[L], t1_buf, HC2_, HC_, 32, g_pb1[L], nullptr, 1);
    k_f32_to_bf16<<<cdiv((long)N_NODES * HC_, 256), 256, 0, stream>>>(bfT, t1_buf, N_NODES * HC_);
    // out = elu(t1 @ phi_W2 + b2 + bias)   (elu follows each layer in reference)
    gemm(bfT, p_pW2[L], out_buf, HC_, HC_, 32, g_pb2[L], g_bias[L], 2);
    // next layer input (bf16)
    k_f32_to_bf16<<<cdiv((long)N_NODES * HC_, 256), 256, 0, stream>>>(bfIn, out_buf, N_NODES * HC_);
  }

  // head: hid = elu(h @ f_W1 + f_b1) ; out = hid @ f_W2 + f_b2
  gemm(bfIn, p_fW1, t1_buf, HC_, HC2_, 64, f_b1, nullptr, 2);
  k_f32_to_bf16<<<cdiv((long)N_NODES * HC2_, 256), 256, 0, stream>>>(bfIn, t1_buf, N_NODES * HC2_);
  gemm(bfIn, p_fW2, (float*)d_out, HC2_, NLABEL_, 3, f_b2, nullptr, 0);
}
